// SparseMoELayer_81922206204349
// MI455X (gfx1250) — matrix operand order repaired and checked by reading, not compiled, not run on hardware
//
#include <hip/hip_runtime.h>
#include <hip/hip_bf16.h>
#include <math.h>

#define N_TOK 8192
#define HDIM  1024
#define FDIM  4096
#define NEXP  8

typedef __bf16 bf16_t;
typedef __bf16 v16bf __attribute__((ext_vector_type(16)));
typedef __bf16 v8bf  __attribute__((ext_vector_type(8)));
typedef __bf16 v4bf  __attribute__((ext_vector_type(4)));
typedef float  v8f   __attribute__((ext_vector_type(8)));

#define BM 128
#define BN 128
#define BK 32
#define LDSTR 40   // bf16 elements per LDS row (pad to dodge bank conflicts, keeps 16B align)

__device__ __forceinline__ v16bf cat8(v8bf lo, v8bf hi) {
    return __builtin_shufflevector(lo, hi, 0,1,2,3,4,5,6,7,8,9,10,11,12,13,14,15);
}

#define WMMA_BF16(A, B, C) __builtin_amdgcn_wmma_f32_16x16x32_bf16( \
        false, (A), false, (B), (short)0, (C), false, false)

// ---------------- convert x to bf16 ----------------
__global__ void k_cvt_x(const float* __restrict__ x, bf16_t* __restrict__ xb, int n) {
    int i = (blockIdx.x * blockDim.x + threadIdx.x) * 4;
    if (i < n) {
        float4 v = *(const float4*)(x + i);
        v4bf o;
        o[0] = (bf16_t)v.x; o[1] = (bf16_t)v.y; o[2] = (bf16_t)v.z; o[3] = (bf16_t)v.w;
        *(v4bf*)(xb + i) = o;
    }
}

// -------- tiled transpose + fp32->bf16 convert: in[E][R][C] -> out[E][C][R] --------
__global__ void k_transpose_cvt(const float* __restrict__ in, bf16_t* __restrict__ out,
                                int R, int C) {
    __shared__ float tile[32][33];
    int e = blockIdx.z;
    const float* src = in + (size_t)e * R * C;
    bf16_t* dst = out + (size_t)e * R * C;
    int c0 = blockIdx.x * 32, r0 = blockIdx.y * 32;
    int tx = threadIdx.x, ty = threadIdx.y;            // 32 x 8
    #pragma unroll
    for (int i = 0; i < 32; i += 8)
        tile[ty + i][tx] = src[(size_t)(r0 + ty + i) * C + c0 + tx];
    __syncthreads();
    #pragma unroll
    for (int i = 0; i < 32; i += 8)
        dst[(size_t)(c0 + ty + i) * R + r0 + tx] = (bf16_t)tile[tx][ty + i];
}

// ---------------- router: one wave per token ----------------
__global__ void k_router(const float* __restrict__ x, const float* __restrict__ grad,
                         const float* __restrict__ Wr, const float* __restrict__ br,
                         int* __restrict__ fill, int* __restrict__ pair_token,
                         float* __restrict__ pair_w) {
    int wave = threadIdx.x >> 5;
    int lane = threadIdx.x & 31;
    int n = blockIdx.x * 8 + wave;
    float acc[NEXP];
    #pragma unroll
    for (int e = 0; e < NEXP; e++) acc[e] = 0.f;
    for (int i = 0; i < HDIM / 32; i++) {
        int h = i * 32 + lane;
        float xv = x[(size_t)n * HDIM + h];
        const float* wr = Wr + (size_t)h * NEXP;
        #pragma unroll
        for (int e = 0; e < NEXP; e++) acc[e] = fmaf(xv, wr[e], acc[e]);
    }
    #pragma unroll
    for (int off = 16; off > 0; off >>= 1)
        #pragma unroll
        for (int e = 0; e < NEXP; e++) acc[e] += __shfl_xor(acc[e], off, 32);
    if (lane == 0) {
        float g = grad[n];
        float logits[NEXP];
        #pragma unroll
        for (int e = 0; e < NEXP; e++)
            logits[e] = acc[e] + g * Wr[(size_t)HDIM * NEXP + e] + br[e];
        float m = logits[0];
        #pragma unroll
        for (int e = 1; e < NEXP; e++) m = fmaxf(m, logits[e]);
        float ex[NEXP], s = 0.f;
        #pragma unroll
        for (int e = 0; e < NEXP; e++) { ex[e] = expf(logits[e] - m); s += ex[e]; }
        float inv = 1.f / s;
        int e1 = 0;
        #pragma unroll
        for (int e = 1; e < NEXP; e++) if (logits[e] > logits[e1]) e1 = e;
        int e2 = (e1 == 0) ? 1 : 0;
        #pragma unroll
        for (int e = 0; e < NEXP; e++)
            if (e != e1 && logits[e] > logits[e2]) e2 = e;
        int s1 = atomicAdd(&fill[e1], 1);
        pair_token[e1 * N_TOK + s1] = n;  pair_w[e1 * N_TOK + s1] = ex[e1] * inv;
        int s2 = atomicAdd(&fill[e2], 1);
        pair_token[e2 * N_TOK + s2] = n;  pair_w[e2 * N_TOK + s2] = ex[e2] * inv;
    }
}

// ---------------- compact h-row base offsets ----------------
__global__ void k_hbase(const int* __restrict__ fill, int* __restrict__ hbase) {
    if (threadIdx.x == 0) {
        int acc = 0;
        for (int e = 0; e < NEXP; e++) {
            hbase[e] = acc;
            acc += ((fill[e] + BM - 1) / BM) * BM;
        }
    }
}

// ---------------- GEMM1: h = gelu(x_gather @ W1 + b1), bf16 out ----------------
__global__ __launch_bounds__(256) void k_ffn1(
    const bf16_t* __restrict__ xb, const bf16_t* __restrict__ W1t,
    const float* __restrict__ b1, const int* __restrict__ fill,
    const int* __restrict__ hbase, const int* __restrict__ pair_token,
    bf16_t* __restrict__ h) {
    int e  = blockIdx.y >> 6;          // 64 row tiles per expert capacity
    int rt = blockIdx.y & 63;
    int cnt = fill[e];
    if (rt * BM >= cnt) return;
    int n0 = blockIdx.x * BN;

    __shared__ bf16_t Als[2][BM * LDSTR];
    __shared__ bf16_t Bls[2][BN * LDSTR];
    __shared__ int toks[BM];

    int t = threadIdx.x;
    if (t < BM) {
        int r = rt * BM + t;
        toks[t] = pair_token[e * N_TOK + ((r < cnt) ? r : 0)];
    }
    __syncthreads();

    int lrow = t >> 1;
    int lseg = (t & 1) * 16;
    const bf16_t* Asrc = xb + (size_t)toks[lrow] * HDIM + lseg;
    const bf16_t* Bsrc = W1t + ((size_t)e * FDIM + n0 + lrow) * HDIM + lseg;

    int wid = t >> 5, lane = t & 31;
    int wm = (wid >> 2) * 64;          // 0 or 64
    int wn = (wid & 3) * 32;           // 0..96
    int l15 = lane & 15, kg = lane >> 4;

    v8f acc[4][2];
    #pragma unroll
    for (int mi = 0; mi < 4; mi++)
        #pragma unroll
        for (int ni = 0; ni < 2; ni++)
            #pragma unroll
            for (int r = 0; r < 8; r++) acc[mi][ni][r] = 0.f;

    // prologue: stage tile 0 into buffer 0
    {
        v8bf a0 = *(const v8bf*)(Asrc);
        v8bf a1 = *(const v8bf*)(Asrc + 8);
        v8bf b0 = *(const v8bf*)(Bsrc);
        v8bf b1v = *(const v8bf*)(Bsrc + 8);
        *(v8bf*)(&Als[0][lrow * LDSTR + lseg])     = a0;
        *(v8bf*)(&Als[0][lrow * LDSTR + lseg + 8]) = a1;
        *(v8bf*)(&Bls[0][lrow * LDSTR + lseg])     = b0;
        *(v8bf*)(&Bls[0][lrow * LDSTR + lseg + 8]) = b1v;
    }
    __syncthreads();

    const int NKT = HDIM / BK;
    for (int it = 0; it < NKT; ++it) {
        int buf = it & 1;
        v8bf pa0, pa1, pb0, pb1;
        if (it + 1 < NKT) {            // prefetch next k-tile into registers
            int kt = (it + 1) * BK;
            pa0 = *(const v8bf*)(Asrc + kt);
            pa1 = *(const v8bf*)(Asrc + kt + 8);
            pb0 = *(const v8bf*)(Bsrc + kt);
            pb1 = *(const v8bf*)(Bsrc + kt + 8);
        }
        v16bf bfrag[2], afrag[4];
        #pragma unroll
        for (int ni = 0; ni < 2; ni++) {
            int brow = wn + ni * 16 + l15;
            bfrag[ni] = cat8(*(const v8bf*)(&Bls[buf][brow * LDSTR + 16 * kg]),
                             *(const v8bf*)(&Bls[buf][brow * LDSTR + 16 * kg + 8]));
        }
        #pragma unroll
        for (int mi = 0; mi < 4; mi++) {
            int arow = wm + mi * 16 + l15;
            afrag[mi] = cat8(*(const v8bf*)(&Als[buf][arow * LDSTR + 16 * kg]),
                             *(const v8bf*)(&Als[buf][arow * LDSTR + 16 * kg + 8]));
        }
        #pragma unroll
        for (int mi = 0; mi < 4; mi++) {
            acc[mi][0] = WMMA_BF16(afrag[mi], bfrag[0], acc[mi][0]);
            acc[mi][1] = WMMA_BF16(afrag[mi], bfrag[1], acc[mi][1]);
        }
        if (it + 1 < NKT) {
            int nb = buf ^ 1;
            *(v8bf*)(&Als[nb][lrow * LDSTR + lseg])     = pa0;
            *(v8bf*)(&Als[nb][lrow * LDSTR + lseg + 8]) = pa1;
            *(v8bf*)(&Bls[nb][lrow * LDSTR + lseg])     = pb0;
            *(v8bf*)(&Bls[nb][lrow * LDSTR + lseg + 8]) = pb1;
        }
        __syncthreads();
    }

    int hrow0 = hbase[e] + rt * BM;
    #pragma unroll
    for (int ni = 0; ni < 2; ni++) {
        int gn = n0 + wn + ni * 16 + l15;
        float bias = b1[(size_t)e * FDIM + gn];
        #pragma unroll
        for (int mi = 0; mi < 4; mi++) {
            #pragma unroll
            for (int r = 0; r < 8; r++) {
                int m = wm + mi * 16 + 8 * kg + r;
                float v = acc[mi][ni][r] + bias;
                float gv = 0.5f * v * (1.0f + erff(v * 0.70710678118654752f));
                h[(size_t)(hrow0 + m) * FDIM + gn] = (bf16_t)gv;
            }
        }
    }
}

// ---------------- GEMM2: out += cw * (h @ W2 + b2), f32 atomics ----------------
__global__ __launch_bounds__(256) void k_ffn2(
    const bf16_t* __restrict__ h, const bf16_t* __restrict__ W2t,
    const float* __restrict__ b2, const int* __restrict__ fill,
    const int* __restrict__ hbase, const int* __restrict__ pair_token,
    const float* __restrict__ pair_w, float* __restrict__ out) {
    int e  = blockIdx.y >> 6;
    int rt = blockIdx.y & 63;
    int cnt = fill[e];
    if (rt * BM >= cnt) return;
    int n0 = blockIdx.x * BN;

    __shared__ bf16_t Als[2][BM * LDSTR];
    __shared__ bf16_t Bls[2][BN * LDSTR];

    int t = threadIdx.x;
    int lrow = t >> 1;
    int lseg = (t & 1) * 16;
    int hrow0 = hbase[e] + rt * BM;
    const bf16_t* Asrc = h + (size_t)(hrow0 + lrow) * FDIM + lseg;
    const bf16_t* Bsrc = W2t + ((size_t)e * HDIM + n0 + lrow) * FDIM + lseg;

    int wid = t >> 5, lane = t & 31;
    int wm = (wid >> 2) * 64;
    int wn = (wid & 3) * 32;
    int l15 = lane & 15, kg = lane >> 4;

    v8f acc[4][2];
    #pragma unroll
    for (int mi = 0; mi < 4; mi++)
        #pragma unroll
        for (int ni = 0; ni < 2; ni++)
            #pragma unroll
            for (int r = 0; r < 8; r++) acc[mi][ni][r] = 0.f;

    {
        v8bf a0 = *(const v8bf*)(Asrc);
        v8bf a1 = *(const v8bf*)(Asrc + 8);
        v8bf b0 = *(const v8bf*)(Bsrc);
        v8bf b1v = *(const v8bf*)(Bsrc + 8);
        *(v8bf*)(&Als[0][lrow * LDSTR + lseg])     = a0;
        *(v8bf*)(&Als[0][lrow * LDSTR + lseg + 8]) = a1;
        *(v8bf*)(&Bls[0][lrow * LDSTR + lseg])     = b0;
        *(v8bf*)(&Bls[0][lrow * LDSTR + lseg + 8]) = b1v;
    }
    __syncthreads();

    const int NKT = FDIM / BK;
    for (int it = 0; it < NKT; ++it) {
        int buf = it & 1;
        v8bf pa0, pa1, pb0, pb1;
        if (it + 1 < NKT) {
            int kt = (it + 1) * BK;
            pa0 = *(const v8bf*)(Asrc + kt);
            pa1 = *(const v8bf*)(Asrc + kt + 8);
            pb0 = *(const v8bf*)(Bsrc + kt);
            pb1 = *(const v8bf*)(Bsrc + kt + 8);
        }
        v16bf bfrag[2], afrag[4];
        #pragma unroll
        for (int ni = 0; ni < 2; ni++) {
            int brow = wn + ni * 16 + l15;
            bfrag[ni] = cat8(*(const v8bf*)(&Bls[buf][brow * LDSTR + 16 * kg]),
                             *(const v8bf*)(&Bls[buf][brow * LDSTR + 16 * kg + 8]));
        }
        #pragma unroll
        for (int mi = 0; mi < 4; mi++) {
            int arow = wm + mi * 16 + l15;
            afrag[mi] = cat8(*(const v8bf*)(&Als[buf][arow * LDSTR + 16 * kg]),
                             *(const v8bf*)(&Als[buf][arow * LDSTR + 16 * kg + 8]));
        }
        #pragma unroll
        for (int mi = 0; mi < 4; mi++) {
            acc[mi][0] = WMMA_BF16(afrag[mi], bfrag[0], acc[mi][0]);
            acc[mi][1] = WMMA_BF16(afrag[mi], bfrag[1], acc[mi][1]);
        }
        if (it + 1 < NKT) {
            int nb = buf ^ 1;
            *(v8bf*)(&Als[nb][lrow * LDSTR + lseg])     = pa0;
            *(v8bf*)(&Als[nb][lrow * LDSTR + lseg + 8]) = pa1;
            *(v8bf*)(&Bls[nb][lrow * LDSTR + lseg])     = pb0;
            *(v8bf*)(&Bls[nb][lrow * LDSTR + lseg + 8]) = pb1;
        }
        __syncthreads();
    }

    float bias[2];
    #pragma unroll
    for (int ni = 0; ni < 2; ni++)
        bias[ni] = b2[(size_t)e * HDIM + n0 + wn + ni * 16 + l15];

    #pragma unroll
    for (int mi = 0; mi < 4; mi++) {
        #pragma unroll
        for (int r = 0; r < 8; r++) {
            int m = wm + mi * 16 + 8 * kg + r;
            int grow = rt * BM + m;
            if (grow < cnt) {
                int tok = pair_token[e * N_TOK + grow];
                float w = pair_w[e * N_TOK + grow];
                #pragma unroll
                for (int ni = 0; ni < 2; ni++) {
                    int gn = n0 + wn + ni * 16 + l15;
                    float v = (acc[mi][ni][r] + bias[ni]) * w;
                    atomicAdd(out + (size_t)tok * HDIM + gn, v);
                }
            }
        }
    }
}

extern "C" void kernel_launch(void* const* d_in, const int* in_sizes, int n_in,
                              void* d_out, int out_size, void* d_ws, size_t ws_size,
                              hipStream_t stream) {
    const float* x    = (const float*)d_in[0];
    const float* grad = (const float*)d_in[1];
    const float* Wr   = (const float*)d_in[2];
    const float* br   = (const float*)d_in[3];
    const float* W1   = (const float*)d_in[4];
    const float* b1   = (const float*)d_in[5];
    const float* W2   = (const float*)d_in[6];
    const float* b2   = (const float*)d_in[7];
    float* out = (float*)d_out;

    char* p = (char*)d_ws;
    const size_t SZ_XB  = (size_t)N_TOK * HDIM * 2;          // 16 MB
    const size_t SZ_W   = (size_t)NEXP * HDIM * FDIM * 2;    // 64 MB each
    const size_t SZ_H   = (size_t)(2 * N_TOK + BM * NEXP) * FDIM * 2;
    const size_t SZ_PT  = (size_t)NEXP * N_TOK * 4;

    bf16_t* xb        = (bf16_t*)p;                 p += SZ_XB;
    bf16_t* W1t       = (bf16_t*)p;                 p += SZ_W;
    bf16_t* W2t       = (bf16_t*)p;                 p += SZ_W;
    bf16_t* hbuf      = (bf16_t*)p;                 p += SZ_H;
    int*    pair_token= (int*)p;                    p += SZ_PT;
    float*  pair_w    = (float*)p;                  p += SZ_PT;
    int*    fill      = (int*)p;                    p += 256;
    int*    hbase     = (int*)p;                    p += 256;

    hipMemsetAsync(out, 0, (size_t)out_size * sizeof(float), stream);
    hipMemsetAsync(fill, 0, NEXP * sizeof(int), stream);

    k_cvt_x<<<(N_TOK * HDIM / 4 + 255) / 256, 256, 0, stream>>>(x, xb, N_TOK * HDIM);
    k_transpose_cvt<<<dim3(FDIM / 32, HDIM / 32, NEXP), dim3(32, 8), 0, stream>>>(W1, W1t, HDIM, FDIM);
    k_transpose_cvt<<<dim3(HDIM / 32, FDIM / 32, NEXP), dim3(32, 8), 0, stream>>>(W2, W2t, FDIM, HDIM);
    k_router<<<N_TOK / 8, 256, 0, stream>>>(x, grad, Wr, br, fill, pair_token, pair_w);
    k_hbase<<<1, 32, 0, stream>>>(fill, hbase);
    k_ffn1<<<dim3(FDIM / BN, NEXP * 64), 256, 0, stream>>>(xb, W1t, b1, fill, hbase, pair_token, hbuf);
    k_ffn2<<<dim3(HDIM / BN, NEXP * 64), 256, 0, stream>>>(hbuf, W2t, b2, fill, hbase, pair_token, pair_w, out);
}
